// ImageEncoder_28999619183135
// MI455X (gfx1250) — matrix operand order repaired and checked by reading, not compiled, not run on hardware
//
#include <hip/hip_runtime.h>
#include <hip/hip_bf16.h>
#include <math.h>

// ---------------------------------------------------------------------------
// Types for CDNA5 WMMA (gfx1250, wave32)
// ---------------------------------------------------------------------------
typedef __bf16 bf16;
typedef __attribute__((ext_vector_type(16))) __bf16 v16bf;
typedef __attribute__((ext_vector_type(4)))  __bf16 v4bf;
typedef __attribute__((ext_vector_type(8)))  float  v8f;

#define WMMA_BF16(A, B, C) \
  __builtin_amdgcn_wmma_f32_16x16x32_bf16(false, (A), false, (B), (short)0, (C), false, false)

__device__ __forceinline__ float gelu_exact(float x) {
  return 0.5f * x * (1.0f + erff(x * 0.70710678118654752f));
}

__device__ __forceinline__ float wave_red_sum(float v) {
#pragma unroll
  for (int off = 16; off > 0; off >>= 1) v += __shfl_xor(v, off, 32);
  return v;
}
// reductions within each 16-lane half of a wave32 (rows of the WMMA C layout)
__device__ __forceinline__ float half_red_max(float v) {
#pragma unroll
  for (int off = 1; off < 16; off <<= 1) v = fmaxf(v, __shfl_xor(v, off, 32));
  return v;
}
__device__ __forceinline__ float half_red_sum(float v) {
#pragma unroll
  for (int off = 1; off < 16; off <<= 1) v += __shfl_xor(v, off, 32);
  return v;
}

// load 8 consecutive floats (2x b128), convert to bf16 into dst[base..base+7];
// `ok==false` zeros the result (select, no branch). Pointer must be legal.
__device__ __forceinline__ void ld8_cvt(v16bf& dst, int base, const float* __restrict__ p,
                                        bool ok) {
  float4 a = *(const float4*)p;
  float4 b = *(const float4*)(p + 4);
  float z = 0.0f;
  float a0 = ok ? a.x : z, a1 = ok ? a.y : z, a2 = ok ? a.z : z, a3 = ok ? a.w : z;
  float b0 = ok ? b.x : z, b1 = ok ? b.y : z, b2 = ok ? b.z : z, b3 = ok ? b.w : z;
  dst[base + 0] = (bf16)a0; dst[base + 1] = (bf16)a1;
  dst[base + 2] = (bf16)a2; dst[base + 3] = (bf16)a3;
  dst[base + 4] = (bf16)b0; dst[base + 5] = (bf16)b1;
  dst[base + 6] = (bf16)b2; dst[base + 7] = (bf16)b3;
}

// load 8 consecutive bf16 from LDS (2x ds_load_b64) into dst[base..base+7]
__device__ __forceinline__ void ld8_lds(v16bf& dst, int base, const bf16* p) {
  v4bf a = *(const v4bf*)p;
  v4bf b = *(const v4bf*)(p + 4);
  dst[base + 0] = a[0]; dst[base + 1] = a[1]; dst[base + 2] = a[2]; dst[base + 3] = a[3];
  dst[base + 4] = b[0]; dst[base + 5] = b[1]; dst[base + 6] = b[2]; dst[base + 7] = b[3];
}

// ---------------------------------------------------------------------------
// WMMA GEMM:  C[M,N] = act( A[M,K] * W[N,K]^T + bias )
//  - block tile 128x128, 256 threads = 8 waves (4x2), wave tile 32x64
//  - 8 x v_wmma_f32_16x16x32_bf16 per wave per 32-deep K step
//  - double-buffered LDS: tile k+1 global loads issue before tile k compute
//  - requires: K % 32 == 0, N % 128 == 0 (true for every call site)
// ---------------------------------------------------------------------------
__global__ __launch_bounds__(256) void k_gemm(
    const float* __restrict__ A, int lda,
    const float* __restrict__ Wt, int ldw,
    const float* __restrict__ bias,
    float* __restrict__ C, int ldc,
    int M, int N, int K, int act) {
  __shared__ bf16 As[2][128][36];
  __shared__ bf16 Bs[2][128][36];

  const int tid  = threadIdx.x;
  const int lane = tid & 31;
  const int wid  = tid >> 5;
  const int wm   = wid & 3;   // 4 wave-rows of 32
  const int wn   = wid >> 2;  // 2 wave-cols of 64
  const int nlo  = lane & 15;
  const int half = lane >> 4;

  const int blockM = blockIdx.y * 128;
  const int blockN = blockIdx.x * 128;

  v8f acc[2][4];
#pragma unroll
  for (int s = 0; s < 2; ++s)
#pragma unroll
    for (int j = 0; j < 4; ++j) acc[s][j] = (v8f){};

  // staging: 128 rows x 32 cols = 1024 float4 per matrix; 4 float4 per thread
  const int srr = tid >> 3;        // base row, +i*32
  const int sc4 = (tid & 7) << 2;  // col 0..28 step 4

  float4 ra[4], rb[4];
  // phase 1: issue all 8 b128 loads (no intervening consumers -> they pipeline)
  auto load_tile = [&](int k0) {
#pragma unroll
    for (int i = 0; i < 4; ++i) {
      int rr = srr + i * 32;
      int gr = blockM + rr;
      int grc = gr < M ? gr : 0;
      ra[i] = *(const float4*)(A + (size_t)grc * lda + k0 + sc4);
      rb[i] = *(const float4*)(Wt + (size_t)(blockN + rr) * ldw + k0 + sc4);
    }
  };
  // phase 2: convert + packed b64 LDS stores
  auto store_tile = [&](int buf) {
#pragma unroll
    for (int i = 0; i < 4; ++i) {
      int rr = srr + i * 32;
      bool ok = (blockM + rr) < M;
      float z = 0.0f;
      v4bf ka;
      ka[0] = (bf16)(ok ? ra[i].x : z); ka[1] = (bf16)(ok ? ra[i].y : z);
      ka[2] = (bf16)(ok ? ra[i].z : z); ka[3] = (bf16)(ok ? ra[i].w : z);
      *(v4bf*)&As[buf][rr][sc4] = ka;
      v4bf kv;
      kv[0] = (bf16)rb[i].x; kv[1] = (bf16)rb[i].y;
      kv[2] = (bf16)rb[i].z; kv[3] = (bf16)rb[i].w;
      *(v4bf*)&Bs[buf][rr][sc4] = kv;
    }
  };

  load_tile(0);
  store_tile(0);
  __syncthreads();
  int buf = 0;

  for (int k0 = 0; k0 < K; k0 += 32) {
    const bool has_next = (k0 + 32) < K;
    if (has_next) load_tile(k0 + 32);  // global loads overlap the WMMAs below

    // ---- fragments from LDS (packed b64 reads) + 8 WMMAs
    v16bf a0, a1;
    {
      const bf16* r0 = &As[buf][wm * 32 + nlo][0];
      const bf16* r1 = &As[buf][wm * 32 + 16 + nlo][0];
      ld8_lds(a0, 0, r0 + half * 8);
      ld8_lds(a0, 8, r0 + 16 + half * 8);
      ld8_lds(a1, 0, r1 + half * 8);
      ld8_lds(a1, 8, r1 + 16 + half * 8);
    }
#pragma unroll
    for (int j = 0; j < 4; ++j) {
      v16bf bm;
      const bf16* br = &Bs[buf][wn * 64 + j * 16 + nlo][0];
      ld8_lds(bm, 0, br + half * 8);
      ld8_lds(bm, 8, br + 16 + half * 8);
      acc[0][j] = WMMA_BF16(a0, bm, acc[0][j]);
      acc[1][j] = WMMA_BF16(a1, bm, acc[1][j]);
    }

    if (has_next) store_tile(buf ^ 1);  // write the *other* buffer
    __syncthreads();
    buf ^= 1;
  }

  // ---- epilogue (N always in-bounds; only M edge guarded)
#pragma unroll
  for (int s = 0; s < 2; ++s) {
    const int rowb = blockM + wm * 32 + s * 16;
#pragma unroll
    for (int j = 0; j < 4; ++j) {
      const int n = blockN + wn * 64 + j * 16 + nlo;
      const float bj = bias ? bias[n] : 0.0f;
#pragma unroll
      for (int r = 0; r < 8; ++r) {
        int m = rowb + r + 8 * half;
        if (m < M) {
          float v = acc[s][j][r] + bj;
          if (act == 1) v = gelu_exact(v);
          C[(size_t)m * ldc + n] = v;
        }
      }
    }
  }
}

// ---------------------------------------------------------------------------
// LayerNorm over last dim (row-major [rows, N], N % 4 == 0); optional add term
// ---------------------------------------------------------------------------
__global__ __launch_bounds__(256) void k_layernorm(
    const float* __restrict__ X, float* __restrict__ Y,
    const float* __restrict__ g, const float* __restrict__ b,
    int N, float eps, const float* __restrict__ add) {
  const int row = blockIdx.x;
  const int N4 = N >> 2;
  const float4* x4 = (const float4*)(X + (size_t)row * N);
  float4* y4 = (float4*)(Y + (size_t)row * N);
  const float4* g4 = (const float4*)g;
  const float4* b4 = (const float4*)b;
  const float4* a4 = add ? (const float4*)(add + (size_t)row * N) : nullptr;

  __shared__ float sred[8];
  __shared__ float smean, srstd;
  const int lane = threadIdx.x & 31, wid = threadIdx.x >> 5;

  float s = 0.0f;
  for (int i = threadIdx.x; i < N4; i += blockDim.x) {
    float4 f = x4[i];
    s += f.x + f.y + f.z + f.w;
  }
  s = wave_red_sum(s);
  if (lane == 0) sred[wid] = s;
  __syncthreads();
  if (threadIdx.x == 0) {
    float t = 0.0f;
#pragma unroll
    for (int i = 0; i < 8; ++i) t += sred[i];
    smean = t / (float)N;
  }
  __syncthreads();
  const float m = smean;
  float v = 0.0f;
  for (int i = threadIdx.x; i < N4; i += blockDim.x) {
    float4 f = x4[i];
    float d0 = f.x - m, d1 = f.y - m, d2 = f.z - m, d3 = f.w - m;
    v += d0 * d0 + d1 * d1 + d2 * d2 + d3 * d3;
  }
  v = wave_red_sum(v);
  if (lane == 0) sred[wid] = v;
  __syncthreads();
  if (threadIdx.x == 0) {
    float t = 0.0f;
#pragma unroll
    for (int i = 0; i < 8; ++i) t += sred[i];
    srstd = rsqrtf(t / (float)N + eps);
  }
  __syncthreads();
  const float rs = srstd;
  for (int i = threadIdx.x; i < N4; i += blockDim.x) {
    float4 f = x4[i];
    float4 gg = g4[i];
    float4 bb = b4[i];
    float4 o;
    o.x = (f.x - m) * rs * gg.x + bb.x;
    o.y = (f.y - m) * rs * gg.y + bb.y;
    o.z = (f.z - m) * rs * gg.z + bb.z;
    o.w = (f.w - m) * rs * gg.w + bb.w;
    if (a4) {
      float4 aa = a4[i];
      o.x += aa.x; o.y += aa.y; o.z += aa.z; o.w += aa.w;
    }
    y4[i] = o;
  }
}

// ---------------------------------------------------------------------------
// Patch-embed im2col: A[4096][768], f = c*256 + iy*16 + ix (float4 along ix)
// ---------------------------------------------------------------------------
__global__ void k_patch_gather(const float* __restrict__ img, float* __restrict__ A) {
  const int p = blockIdx.x;
  const int py = p >> 6, px = p & 63;
  float4* dst = (float4*)(A + (size_t)p * 768);
  for (int f4 = threadIdx.x; f4 < 192; f4 += blockDim.x) {
    int f = f4 << 2;
    int c = f >> 8, rem = f & 255, iy = rem >> 4, ix = rem & 15;
    const float4* src =
        (const float4*)(img + ((size_t)c * 1024 + py * 16 + iy) * 1024 + px * 16 + ix);
    dst[f4] = *src;
  }
}

// ---------------------------------------------------------------------------
// Window partition (zero pad): X[4096,768] -> Y[nW*S*S, 768]; global: S=64,nWx=1
// ---------------------------------------------------------------------------
__global__ void k_window_gather(const float* __restrict__ X, float* __restrict__ Y,
                                int S, int nWx) {
  const int t = blockIdx.x;
  const int Lw = S * S;
  const int w = t / Lw, l = t - w * Lw;
  const int wy = w / nWx, wx = w - wy * nWx;
  const int ly = l / S, lx = l - ly * S;
  const int gy = wy * S + ly, gx = wx * S + lx;
  const bool ok = (gy < 64) && (gx < 64);
  const int gp = ok ? (gy * 64 + gx) : 0;
  const float4* src = (const float4*)(X + (size_t)gp * 768);
  float4* dst = (float4*)(Y + (size_t)t * 768);
  const float4 zz = make_float4(0.f, 0.f, 0.f, 0.f);
  for (int c = threadIdx.x; c < 192; c += blockDim.x) {
    float4 f = src[c];
    dst[c] = ok ? f : zz;
  }
}

// Window unpartition + residual add: Out[4096,768] = unpart(Yw) + Xres
__global__ void k_unpart_add(const float* __restrict__ Yw, const float* __restrict__ Xres,
                             float* __restrict__ Out, int S, int nWx) {
  const int p = blockIdx.x;
  const int gy = p >> 6, gx = p & 63;
  const int wy = gy / S, wx = gx / S;
  const int ly = gy - wy * S, lx = gx - wx * S;
  const int t = (wy * nWx + wx) * S * S + ly * S + lx;
  const float4* src = (const float4*)(Yw + (size_t)t * 768);
  const float4* res = (const float4*)(Xres + (size_t)p * 768);
  float4* dst = (float4*)(Out + (size_t)p * 768);
  for (int c = threadIdx.x; c < 192; c += blockDim.x) {
    float4 a = src[c], b = res[c];
    dst[c] = make_float4(a.x + b.x, a.y + b.y, a.z + b.z, a.w + b.w);
  }
}

// ---------------------------------------------------------------------------
// Scatter qkv GEMM output into q/k/v tensors [(bw*12+h)][l][d]
// channel c: d = c/36, h = (c/3)%12, which = c%3
// ---------------------------------------------------------------------------
__global__ void k_qkv_scatter(const float* __restrict__ QKV,
                              float* __restrict__ Qb, float* __restrict__ Kb,
                              float* __restrict__ Vb, int Lw, size_t total) {
  size_t e = (size_t)blockIdx.x * blockDim.x + threadIdx.x;
  if (e >= total) return;
  int c = (int)(e % 2304);
  size_t t = e / 2304;
  int l = (int)(t % Lw);
  int bw = (int)(t / Lw);
  int d = c / 36;
  int h = (c / 3) % 12;
  int which = c % 3;
  size_t dst = (((size_t)(bw * 12 + h)) * Lw + l) * 64 + d;
  float v = QKV[e];
  if (which == 0) Qb[dst] = v;
  else if (which == 1) Kb[dst] = v;
  else Vb[dst] = v;
}

// ---------------------------------------------------------------------------
// Rel-pos tables: Out[bh][q][kpos] = dot64(Q[bh][q], relTab[qpos-kpos+S-1])
//  mode 0: qpos = q / S (rel_h), mode 1: qpos = q % S (rel_w). H == W == S.
// ---------------------------------------------------------------------------
__global__ void k_relpos(const float* __restrict__ Qb, const float* __restrict__ relTab,
                         float* __restrict__ Out, int L, int S, int nBH, int mode,
                         size_t total) {
  size_t e = (size_t)blockIdx.x * blockDim.x + threadIdx.x;
  if (e >= total) return;
  int kpos = (int)(e % S);
  size_t t = e / S;
  int q = (int)(t % L);
  int bh = (int)(t / L);
  int qpos = mode ? (q % S) : (q / S);
  const float4* qv = (const float4*)(Qb + ((size_t)bh * L + q) * 64);
  const float4* rv = (const float4*)(relTab + (size_t)(qpos - kpos + S - 1) * 64);
  float s = 0.0f;
#pragma unroll
  for (int d = 0; d < 16; ++d) {
    float4 a = qv[d], b = rv[d];
    s += a.x * b.x + a.y * b.y + a.z * b.z + a.w * b.w;
  }
  Out[e] = s;
}

// ---------------------------------------------------------------------------
// Flash attention chunk: FULL=true -> no bounds handling at all (wave-uniform
// fast path); FULL=false -> clamped addresses + selects for the final chunk.
// ---------------------------------------------------------------------------
template <bool FULL>
__device__ __forceinline__ void flash_chunk(
    int kc, int L, int S, int bh, int q0, int nlo, int half,
    const float* __restrict__ Kb, const float* __restrict__ Vb,
    const float* __restrict__ relH, const float* __restrict__ relW,
    const v16bf& qa0, const v16bf& qa1,
    v8f& o0, v8f& o1, v8f& o2, v8f& o3,
    float* mrow, float* lrow, bf16 (*Pz)[36]) {
  const float scale = 0.125f;  // 64^-0.5
  v8f s0 = {}, s1 = {};
  {
    const int n0 = kc + nlo, n1 = kc + 16 + nlo;
    const bool ok0 = FULL ? true : (n0 < L);
    const bool ok1 = FULL ? true : (n1 < L);
    const float* k0p = Kb + (size_t)(ok0 ? n0 : 0) * 64;
    const float* k1p = Kb + (size_t)(ok1 ? n1 : 0) * 64;
    v16bf kb00, kb01, kb10, kb11;
    ld8_cvt(kb00, 0, k0p + half * 16, ok0);
    ld8_cvt(kb00, 8, k0p + half * 16 + 8, ok0);
    ld8_cvt(kb01, 0, k0p + 32 + half * 16, ok0);
    ld8_cvt(kb01, 8, k0p + 32 + half * 16 + 8, ok0);
    ld8_cvt(kb10, 0, k1p + half * 16, ok1);
    ld8_cvt(kb10, 8, k1p + half * 16 + 8, ok1);
    ld8_cvt(kb11, 0, k1p + 32 + half * 16, ok1);
    ld8_cvt(kb11, 8, k1p + 32 + half * 16 + 8, ok1);
    s0 = WMMA_BF16(qa0, kb00, s0);
    s0 = WMMA_BF16(qa1, kb01, s0);
    s1 = WMMA_BF16(qa0, kb10, s1);
    s1 = WMMA_BF16(qa1, kb11, s1);
  }

  // scale + rel-pos (+ mask on tail); kh/kw hoisted out of the row loop
  {
    const int key0 = kc + nlo, key1 = kc + 16 + nlo;
    const int k0c = FULL ? key0 : min(key0, L - 1);
    const int k1c = FULL ? key1 : min(key1, L - 1);
    const int kh0 = k0c / S, kw0 = k0c - kh0 * S;
    const int kh1 = k1c / S, kw1 = k1c - kh1 * S;
    const bool kv0 = FULL ? true : (key0 < L);
    const bool kv1 = FULL ? true : (key1 < L);
#pragma unroll
    for (int r = 0; r < 8; ++r) {
      int q = q0 + r + 8 * half;
      int qc = min(q, L - 1);
      size_t rb = ((size_t)bh * L + qc) * S;
      float rh0 = relH[rb + kh0], rw0 = relW[rb + kw0];
      float rh1 = relH[rb + kh1], rw1 = relW[rb + kw1];
      bool okq = q < L;
      float v0 = s0[r] * scale + rh0 + rw0;
      float v1 = s1[r] * scale + rh1 + rw1;
      s0[r] = (okq && kv0) ? v0 : -1e30f;
      s1[r] = (okq && kv1) ? v1 : -1e30f;
    }
  }

  // online softmax over the 32-key chunk
  float pscale[8];
#pragma unroll
  for (int r = 0; r < 8; ++r) {
    float mx = half_red_max(fmaxf(s0[r], s1[r]));
    float mnew = fmaxf(mrow[r], mx);
    float corr = __expf(mrow[r] - mnew);
    float p0 = __expf(s0[r] - mnew);
    float p1 = __expf(s1[r] - mnew);
    float ps = half_red_sum(p0 + p1);
    lrow[r] = lrow[r] * corr + ps;
    mrow[r] = mnew;
    pscale[r] = corr;
    s0[r] = p0;
    s1[r] = p1;
  }
#pragma unroll
  for (int r = 0; r < 8; ++r) {
    o0[r] *= pscale[r]; o1[r] *= pscale[r];
    o2[r] *= pscale[r]; o3[r] *= pscale[r];
  }

  // P (C layout) -> LDS -> A fragment layout (packed b64 reads)
#pragma unroll
  for (int r = 0; r < 8; ++r) {
    Pz[r + 8 * half][nlo]      = (bf16)s0[r];
    Pz[r + 8 * half][nlo + 16] = (bf16)s1[r];
  }
  __syncthreads();
  v16bf pa;
  {
    const bf16* pr = &Pz[nlo][0];
    ld8_lds(pa, 0, pr + half * 16);
    ld8_lds(pa, 8, pr + half * 16 + 8);
  }
  __syncthreads();

  // V fragments (column gather) and accumulate O
#pragma unroll
  for (int j = 0; j < 4; ++j) {
    v16bf vbF;
    const int col = j * 16 + nlo;
#pragma unroll
    for (int g = 0; g < 8; ++g) {
      int kk = kc + half * 16 + 2 * g;
      float e0, e1;
      if (FULL) {
        e0 = Vb[(size_t)kk * 64 + col];
        e1 = Vb[(size_t)(kk + 1) * 64 + col];
      } else {
        int ka = min(kk, L - 1), kb2 = min(kk + 1, L - 1);
        e0 = Vb[(size_t)ka * 64 + col];
        e1 = Vb[(size_t)kb2 * 64 + col];
        e0 = (kk < L) ? e0 : 0.0f;
        e1 = (kk + 1 < L) ? e1 : 0.0f;
      }
      vbF[2 * g]     = (bf16)e0;
      vbF[2 * g + 1] = (bf16)e1;
    }
    if (j == 0) o0 = WMMA_BF16(pa, vbF, o0);
    else if (j == 1) o1 = WMMA_BF16(pa, vbF, o1);
    else if (j == 2) o2 = WMMA_BF16(pa, vbF, o2);
    else o3 = WMMA_BF16(pa, vbF, o3);
  }
}

// ---------------------------------------------------------------------------
// Flash attention: one wave32 per (bh, 16-query tile)
// ---------------------------------------------------------------------------
__global__ __launch_bounds__(32) void k_flash_attn(
    const float* __restrict__ Q, const float* __restrict__ K,
    const float* __restrict__ V,
    const float* __restrict__ relH, const float* __restrict__ relW,
    float* __restrict__ Out, int L, int S, int heads) {
  const int bh = blockIdx.y;
  const int bw = bh / heads;
  const int head = bh - bw * heads;
  const int q0 = blockIdx.x * 16;
  const int lane = threadIdx.x;
  const int nlo = lane & 15;
  const int half = lane >> 4;

  const float* Qb = Q + (size_t)bh * L * 64;
  const float* Kb = K + (size_t)bh * L * 64;
  const float* Vb = V + (size_t)bh * L * 64;

  // Q fragments: contiguous 8-float runs of the query row
  v16bf qa0, qa1;
  {
    const int m = q0 + nlo;
    const bool mok = m < L;
    const float* qr = Qb + (size_t)(mok ? m : 0) * 64;
    ld8_cvt(qa0, 0, qr + half * 16, mok);
    ld8_cvt(qa0, 8, qr + half * 16 + 8, mok);
    ld8_cvt(qa1, 0, qr + 32 + half * 16, mok);
    ld8_cvt(qa1, 8, qr + 32 + half * 16 + 8, mok);
  }

  float mrow[8], lrow[8];
#pragma unroll
  for (int r = 0; r < 8; ++r) { mrow[r] = -1e30f; lrow[r] = 0.0f; }
  v8f o0 = {}, o1 = {}, o2 = {}, o3 = {};

  __shared__ bf16 Pz[16][36];

  const int kfull = L & ~31;  // keys covered by full 32-chunks
  for (int kc = 0; kc < kfull; kc += 32)
    flash_chunk<true>(kc, L, S, bh, q0, nlo, half, Kb, Vb, relH, relW,
                      qa0, qa1, o0, o1, o2, o3, mrow, lrow, Pz);
  if (kfull < L)
    flash_chunk<false>(kfull, L, S, bh, q0, nlo, half, Kb, Vb, relH, relW,
                       qa0, qa1, o0, o1, o2, o3, mrow, lrow, Pz);

  // normalize and store head-merged: Out[bw][q][head*64 + d]
#pragma unroll
  for (int r = 0; r < 8; ++r) {
    int q = q0 + r + 8 * half;
    if (q >= L) continue;
    float inv = 1.0f / lrow[r];
    size_t base = ((size_t)bw * L + q) * (size_t)(heads * 64) + (size_t)head * 64;
    Out[base + 0 + nlo]  = o0[r] * inv;
    Out[base + 16 + nlo] = o1[r] * inv;
    Out[base + 32 + nlo] = o2[r] * inv;
    Out[base + 48 + nlo] = o3[r] * inv;
  }
}

// ---------------------------------------------------------------------------
// Neck helpers
// ---------------------------------------------------------------------------
__global__ void k_im2col3(const float* __restrict__ X, float* __restrict__ A) {
  const int p = blockIdx.x;
  const int y = p >> 6, x = p & 63;
  for (int f = threadIdx.x; f < 2304; f += blockDim.x) {
    int c = f / 9, rem = f - c * 9;
    int dy = rem / 3 - 1, dx = rem % 3 - 1;
    int yy = y + dy, xx = x + dx;
    bool ok = (yy >= 0 && yy < 64 && xx >= 0 && xx < 64);
    int src = ok ? (yy * 64 + xx) : 0;
    float v = X[(size_t)src * 256 + c];
    A[(size_t)p * 2304 + f] = ok ? v : 0.0f;
  }
}

__global__ void k_out_nchw(const float* __restrict__ X, float* __restrict__ Out) {
  size_t e = (size_t)blockIdx.x * blockDim.x + threadIdx.x;
  if (e >= (size_t)4096 * 256) return;
  int p = (int)(e >> 8), c = (int)(e & 255);
  Out[(size_t)c * 4096 + p] = X[e];
}

// ---------------------------------------------------------------------------
// Host-side launch
// ---------------------------------------------------------------------------
extern "C" void kernel_launch(void* const* d_in, const int* in_sizes, int n_in,
                              void* d_out, int out_size, void* d_ws, size_t ws_size,
                              hipStream_t stream) {
  (void)in_sizes; (void)n_in; (void)out_size; (void)ws_size;
  // Input order (setup_inputs dict order, nested insertion order):
  // 0:x 1:patch_w 2:patch_b 3:pn_g 4:pn_b 5:ape 6:c1_w 7:ln1_g 8:ln1_b
  // 9:c2_w 10:ln2_g 11:ln2_b, then 4 blocks of 16 params starting at 12.
  const float* x_img   = (const float*)d_in[0];
  const float* patch_w = (const float*)d_in[1];
  const float* patch_b = (const float*)d_in[2];
  const float* pn_g    = (const float*)d_in[3];
  const float* pn_b    = (const float*)d_in[4];
  const float* ape     = (const float*)d_in[5];
  const float* c1_w    = (const float*)d_in[6];
  const float* ln1_g   = (const float*)d_in[7];
  const float* ln1_b   = (const float*)d_in[8];
  const float* c2_w    = (const float*)d_in[9];
  const float* ln2_g   = (const float*)d_in[10];
  const float* ln2_b   = (const float*)d_in[11];

  // workspace linear allocator (floats)
  float* wsf = (float*)d_ws;
  size_t off = 0;
  auto alloc = [&](size_t n) { float* p = wsf + off; off += n; return p; };
  float* tokA  = alloc((size_t)4096 * 768);
  float* tokB  = alloc((size_t)4096 * 768);
  float* xw    = alloc((size_t)4900 * 768);
  float* qkvb  = alloc((size_t)4900 * 2304);
  float* qb    = alloc((size_t)4900 * 768);
  float* kb    = alloc((size_t)4900 * 768);
  float* vb    = alloc((size_t)4900 * 768);
  float* relHb = alloc((size_t)12 * 4096 * 64);
  float* relWb = alloc((size_t)12 * 4096 * 64);
  float* attnO = alloc((size_t)4900 * 768);
  float* projb = alloc((size_t)4900 * 768);
  float* x2    = alloc((size_t)4096 * 768);
  float* z1    = alloc((size_t)4096 * 768);
  float* z2    = alloc((size_t)4096 * 768);
  float* hbuf  = alloc((size_t)4096 * 3072);
  float* nim   = alloc((size_t)4096 * 2304);
  float* nb1   = alloc((size_t)4096 * 256);
  float* nb2   = alloc((size_t)4096 * 256);

  const dim3 T256(256);

  // ---- patch embed: im2col + GEMM(4096x768x768) + bias, then LN(pn) + ape
  k_patch_gather<<<4096, T256, 0, stream>>>(x_img, xw);
  k_gemm<<<dim3(768 / 128, 4096 / 128), T256, 0, stream>>>(
      xw, 768, patch_w, 768, patch_b, tokA, 768, 4096, 768, 768, 0);
  k_layernorm<<<4096, T256, 0, stream>>>(tokA, tokB, pn_g, pn_b, 768, 1e-6f, ape);

  float* tok = tokB;
  float* tokN = tokA;

  static const int ws_list[4] = {14, 14, 14, 0};
  for (int blk = 0; blk < 4; ++blk) {
    const float* const* P = (const float* const*)(d_in + 12 + blk * 16);
    const float* n1_g = P[0];   const float* n1_b = P[1];
    const float* qkv_w = P[2];  const float* qkv_b = P[3];
    const float* rel_h = P[4];  const float* rel_w = P[5];
    const float* proj_w = P[6]; const float* proj_b = P[7];
    const float* n2_g = P[8];   const float* n2_b = P[9];
    const float* mn_g = P[10];  const float* mn_b = P[11];
    const float* fc1_w = P[12]; const float* fc1_b = P[13];
    const float* fc2_w = P[14]; const float* fc2_b = P[15];

    const int ws = ws_list[blk];
    const int S   = (ws > 0) ? ws : 64;  // window (= attention H = W)
    const int nWx = (ws > 0) ? 5 : 1;    // ceil(64/14)=5; global: 1
    const int Lw  = S * S;
    const int nW  = nWx * nWx;
    const int Mw  = nW * Lw;
    const int nBH = nW * 12;

    // 1) LN(n1)
    k_layernorm<<<4096, T256, 0, stream>>>(tok, z1, n1_g, n1_b, 768, 1e-6f, nullptr);
    // 2) window partition (zero-padded)
    k_window_gather<<<Mw, T256, 0, stream>>>(z1, xw, S, nWx);
    // 3) qkv projection
    k_gemm<<<dim3(2304 / 128, (Mw + 127) / 128), T256, 0, stream>>>(
        xw, 768, qkv_w, 768, qkv_b, qkvb, 2304, Mw, 2304, 768, 0);
    // 4) scatter into per-head q/k/v
    {
      size_t total = (size_t)Mw * 2304;
      k_qkv_scatter<<<(unsigned)((total + 255) / 256), T256, 0, stream>>>(
          qkvb, qb, kb, vb, Lw, total);
    }
    // 5) decomposed rel-pos tables
    {
      size_t total = (size_t)nBH * Lw * S;
      unsigned g = (unsigned)((total + 255) / 256);
      k_relpos<<<g, T256, 0, stream>>>(qb, rel_h, relHb, Lw, S, nBH, 0, total);
      k_relpos<<<g, T256, 0, stream>>>(qb, rel_w, relWb, Lw, S, nBH, 1, total);
    }
    // 6) flash attention (WMMA), head-merged output
    k_flash_attn<<<dim3((Lw + 15) / 16, nBH), dim3(32), 0, stream>>>(
        qb, kb, vb, relHb, relWb, attnO, Lw, S, 12);
    // 7) output projection
    k_gemm<<<dim3(768 / 128, (Mw + 127) / 128), T256, 0, stream>>>(
        attnO, 768, proj_w, 768, proj_b, projb, 768, Mw, 768, 768, 0);
    // 8) unpartition + residual
    k_unpart_add<<<4096, T256, 0, stream>>>(projb, tok, x2, S, nWx);
    // 9) double norm (n2 then mlp-internal norm)
    k_layernorm<<<4096, T256, 0, stream>>>(x2, z1, n2_g, n2_b, 768, 1e-6f, nullptr);
    k_layernorm<<<4096, T256, 0, stream>>>(z1, z2, mn_g, mn_b, 768, 1e-12f, nullptr);
    // 10) MLP: fc1 + exact GELU, fc2 (NO second residual per reference)
    k_gemm<<<dim3(3072 / 128, 4096 / 128), T256, 0, stream>>>(
        z2, 768, fc1_w, 768, fc1_b, hbuf, 3072, 4096, 3072, 768, 1);
    k_gemm<<<dim3(768 / 128, 4096 / 128), T256, 0, stream>>>(
        hbuf, 3072, fc2_w, 3072, fc2_b, tokN, 768, 4096, 768, 3072, 0);

    float* t = tok; tok = tokN; tokN = t;
  }

  // ---- neck: 1x1 conv -> LN2d -> 3x3 conv -> LN2d -> NCHW
  k_gemm<<<dim3(256 / 128, 4096 / 128), T256, 0, stream>>>(
      tok, 768, c1_w, 768, nullptr, nb1, 256, 4096, 256, 768, 0);
  k_layernorm<<<4096, T256, 0, stream>>>(nb1, nb2, ln1_g, ln1_b, 256, 1e-6f, nullptr);
  k_im2col3<<<4096, T256, 0, stream>>>(nb2, nim);
  k_gemm<<<dim3(256 / 128, 4096 / 128), T256, 0, stream>>>(
      nim, 2304, c2_w, 2304, nullptr, nb1, 256, 4096, 256, 2304, 0);
  k_layernorm<<<4096, T256, 0, stream>>>(nb1, nb2, ln2_g, ln2_b, 256, 1e-6f, nullptr);
  k_out_nchw<<<4096, T256, 0, stream>>>(nb2, (float*)d_out);
}
